// NeuronGPTOSSBlock_6691559047326
// MI455X (gfx1250) — matrix operand order repaired and checked by reading, not compiled, not run on hardware
//
#include <hip/hip_runtime.h>
#include <hip/hip_bf16.h>
#include <math.h>

typedef _Float16 half_t;
typedef __attribute__((ext_vector_type(16))) _Float16 v16h;
typedef __attribute__((ext_vector_type(8)))  _Float16 v8h;
typedef __attribute__((ext_vector_type(8)))  float    v8f;

constexpr int Bc  = 2;
constexpr int Sc  = 2048;
constexpr int Hc  = 2048;
constexpr int NHc = 16;
constexpr int HDc = 128;
constexpr int Ec  = 8;
constexpr int Ic  = 2048;
constexpr int Tc  = Bc * Sc;
constexpr float EPSc = 1e-5f;
constexpr int BKP = 40;  // padded LDS row stride in halves (32 data + 8 pad)

#define DEVINL __device__ __forceinline__

// ---------------------------------------------------------------------------
// CDNA5 async copy global->LDS (ASYNCcnt-tracked). LDS address = low 32 bits
// of the flat pointer (ISA 10.2: LDS aperture maps addr[31:0] -> LDS offset).
// ---------------------------------------------------------------------------
DEVINL void async_ld_b128(unsigned lds_off, const half_t* g) {
  asm volatile("global_load_async_to_lds_b128 %0, %1, off"
               :: "v"(lds_off), "v"((unsigned long long)(uintptr_t)g)
               : "memory");
}
DEVINL void wait_async0() {
  asm volatile("s_wait_asynccnt 0x0" ::: "memory");
}
DEVINL unsigned lds_off(const void* p) { return (unsigned)(uintptr_t)p; }

// ---------------------------------------------------------------------------
// WMMA fragment helpers (wave32, V_WMMA_F32_16X16X32_F16 layouts, ISA 7.12.2)
// ---------------------------------------------------------------------------
DEVINL v16h frag_cat(v8h lo, v8h hi) {
  v16h r;
#pragma unroll
  for (int i = 0; i < 8; ++i) { r[i] = lo[i]; r[i + 8] = hi[i]; }
  return r;
}

// A fragment 16x32 (MxK): lane<16 row=lane holds K {kb+0..7, kb+16..23};
// lane>=16 row=lane-16 holds K {kb+8..15, kb+24..31}. rowptr=&A[row][0].
DEVINL v16h load_a_frag(const half_t* rowptr, int kbase, int lane) {
  const half_t* p = rowptr + kbase + ((lane & 16) ? 8 : 0);
  return frag_cat(*(const v8h*)p, *(const v8h*)(p + 16));
}

// B fragment 32x16 (KxN) from B^T stored row-major [N][K]:
// lane<16: n=lane, K {kb+0..7, kb+16..23}; lane>=16: n=lane-16, K {kb+8..15, kb+24..31}.
DEVINL v16h load_b_frag(const half_t* rowptr, int kbase, int lane) {
  const half_t* p = rowptr + kbase + ((lane & 16) ? 8 : 0);
  return frag_cat(*(const v8h*)p, *(const v8h*)(p + 16));
}

DEVINL v8f wmma_f16(v16h a, v16h b, v8f c) {
  return __builtin_amdgcn_wmma_f32_16x16x32_f16(false, a, false, b, (short)0, c,
                                                false, false);
}

DEVINL float lane16_max(float v) {
#pragma unroll
  for (int m = 1; m < 16; m <<= 1) v = fmaxf(v, __shfl_xor(v, m, 32));
  return v;
}
DEVINL float lane16_sum(float v) {
#pragma unroll
  for (int m = 1; m < 16; m <<= 1) v += __shfl_xor(v, m, 32);
  return v;
}
DEVINL float wave_sum(float v) {
#pragma unroll
  for (int m = 1; m < 32; m <<= 1) v += __shfl_xor(v, m, 32);
  return v;
}

// ---------------------------------------------------------------------------
// Weight transpose + f32->f16 cast: src [R][C] f32 -> dst [C][R] f16, batch z
// ---------------------------------------------------------------------------
__global__ void transpose_cast_kernel(const float* __restrict__ src,
                                      half_t* __restrict__ dst, int R, int C) {
  int r = blockIdx.x * 16 + threadIdx.x;
  int c = blockIdx.y * 16 + threadIdx.y;
  size_t bs = (size_t)blockIdx.z * (size_t)R * (size_t)C;
  if (r < R && c < C)
    dst[bs + (size_t)c * R + r] = (half_t)src[bs + (size_t)r * C + c];
}

// ---------------------------------------------------------------------------
// RMSNorm: one block per token row; writes f16
// ---------------------------------------------------------------------------
__global__ __launch_bounds__(256) void rmsnorm_kernel(
    const float* __restrict__ x, const float* __restrict__ w,
    half_t* __restrict__ out16, int H) {
  int tok = blockIdx.x;
  const float* xr = x + (size_t)tok * H;
  int lane = threadIdx.x & 31, wave = threadIdx.x >> 5;
  __shared__ float red[8];
  float s = 0.f;
  for (int k = threadIdx.x; k < H; k += 256) { float v = xr[k]; s += v * v; }
  s = wave_sum(s);
  if (lane == 0) red[wave] = s;
  __syncthreads();
  if (wave == 0) {
    float t = (lane < 8) ? red[lane] : 0.f;
#pragma unroll
    for (int m = 1; m < 8; m <<= 1) t += __shfl_xor(t, m, 32);
    if (lane == 0) red[0] = rsqrtf(t / (float)H + EPSc);
  }
  __syncthreads();
  float inv = red[0];
  for (int k = threadIdx.x; k < H; k += 256)
    out16[(size_t)tok * H + k] = (half_t)(xr[k] * inv * w[k]);
}

// ---------------------------------------------------------------------------
// Tiled GEMM with double-buffered async global->LDS staging.
// C[M][N] f32 = A[M][K] f16 x Bt[N][K] f16.
// 256 threads = 8 waves; block tile 64(M) x 128(N) x 32(K);
// wave (wm=w&3, wn=w>>2) owns a 16x64 strip: 4 accumulators.
// ---------------------------------------------------------------------------
__global__ __launch_bounds__(256) void gemm_lds_kernel(
    const half_t* __restrict__ A, const half_t* __restrict__ Bt,
    float* __restrict__ C, int M, int N, int K) {
  __shared__ half_t sA[2][64 * BKP];
  __shared__ half_t sB[2][128 * BKP];
  int tid = threadIdx.x, lane = tid & 31, wave = tid >> 5;
  int wm = wave & 3, wn = wave >> 2;
  int mblk = blockIdx.x * 64, nblk = blockIdx.y * 128;

  // fixed per-thread staging slots (16B chunks): A 256 chunks, B 512 chunks
  int arow = tid >> 2, aseg = (tid & 3) * 8;
  const half_t* aSrc = A + (size_t)(mblk + arow) * K + aseg;
  unsigned aDst[2] = {lds_off(&sA[0][arow * BKP + aseg]),
                      lds_off(&sA[1][arow * BKP + aseg])};
  const half_t* bSrc[2];
  unsigned bDst[2][2];
#pragma unroll
  for (int i = 0; i < 2; ++i) {
    int c = tid + i * 256, brow = c >> 2, bseg = (c & 3) * 8;
    bSrc[i] = Bt + (size_t)(nblk + brow) * K + bseg;
    bDst[0][i] = lds_off(&sB[0][brow * BKP + bseg]);
    bDst[1][i] = lds_off(&sB[1][brow * BKP + bseg]);
  }
  auto stage = [&](int buf, int k0) {
    async_ld_b128(aDst[buf], aSrc + k0);
    async_ld_b128(bDst[buf][0], bSrc[0] + k0);
    async_ld_b128(bDst[buf][1], bSrc[1] + k0);
  };

  v8f zero = {};
  v8f acc[4] = {zero, zero, zero, zero};
  int ksteps = K >> 5;
  stage(0, 0);
  wait_async0();
  __syncthreads();
  for (int ks = 0; ks < ksteps; ++ks) {
    if (ks + 1 < ksteps) stage((ks + 1) & 1, (ks + 1) * 32);
    const half_t* aL = &sA[ks & 1][(wm * 16 + (lane & 15)) * BKP];
    v16h af = load_a_frag(aL, 0, lane);
#pragma unroll
    for (int nc = 0; nc < 4; ++nc) {
      const half_t* bL = &sB[ks & 1][(wn * 64 + nc * 16 + (lane & 15)) * BKP];
      acc[nc] = wmma_f16(af, load_b_frag(bL, 0, lane), acc[nc]);
    }
    wait_async0();
    __syncthreads();
  }
  int n0 = nblk + wn * 64 + (lane & 15);
  int m0 = mblk + wm * 16 + ((lane & 16) ? 8 : 0);
#pragma unroll
  for (int nc = 0; nc < 4; ++nc)
#pragma unroll
    for (int j = 0; j < 8; ++j)
      C[(size_t)(m0 + j) * N + n0 + nc * 16] = acc[nc][j];
}

// ---------------------------------------------------------------------------
// RoPE + permute [T][H] f32 -> [B][NH][S][HD] f16 (applied to q and k)
// ---------------------------------------------------------------------------
__global__ __launch_bounds__(64) void rope_permute_kernel(
    const float* __restrict__ qf, const float* __restrict__ kf,
    const int* __restrict__ pos_ids, half_t* __restrict__ q16,
    half_t* __restrict__ k16) {
  int t = blockIdx.x, h = blockIdx.y, d = threadIdx.x;  // d in [0,64)
  int b = t / Sc, s = t % Sc;
  float pos = (float)pos_ids[t];
  float inv_freq = powf(10000.f, -((float)d / 64.f));
  float ang = pos * inv_freq, sn, cs;
  sincosf(ang, &sn, &cs);
  size_t src = (size_t)t * Hc + (size_t)h * HDc + d;
  size_t dst = ((size_t)(b * NHc + h) * Sc + s) * HDc + d;
  float x1 = qf[src], x2 = qf[src + 64];
  q16[dst]      = (half_t)(x1 * cs - x2 * sn);
  q16[dst + 64] = (half_t)(x2 * cs + x1 * sn);
  x1 = kf[src]; x2 = kf[src + 64];
  k16[dst]      = (half_t)(x1 * cs - x2 * sn);
  k16[dst + 64] = (half_t)(x2 * cs + x1 * sn);
}

__global__ __launch_bounds__(128) void v_permute_kernel(
    const float* __restrict__ vf, half_t* __restrict__ v16) {
  int t = blockIdx.x, h = blockIdx.y, d = threadIdx.x;
  int b = t / Sc, s = t % Sc;
  v16[((size_t)(b * NHc + h) * Sc + s) * HDc + d] =
      (half_t)vf[(size_t)t * Hc + (size_t)h * HDc + d];
}

// ---------------------------------------------------------------------------
// Flash attention: one wave per (b, h, 16-row q tile). 32-key blocks,
// causal mask, online softmax, WMMA for QK^T and PV.
// ---------------------------------------------------------------------------
__global__ __launch_bounds__(32) void flash_attn_kernel(
    const half_t* __restrict__ q16, const half_t* __restrict__ k16,
    const half_t* __restrict__ v16, half_t* __restrict__ ctx16) {
  int lane = threadIdx.x & 31;
  int qt = blockIdx.x, h = blockIdx.y, b = blockIdx.z;
  int qbase = qt * 16;
  size_t bh = (size_t)(b * NHc + h) * Sc;
  const half_t* Q = q16 + bh * HDc;
  const half_t* K = k16 + bh * HDc;
  const half_t* V = v16 + bh * HDc;

  __shared__ half_t sP[16 * 32];    // P tile (C-layout -> A-layout shuffle)
  __shared__ half_t sVt[HDc * 32];  // V^T tile: sVt[d*32 + kk]

  v16h aq[4];
  const half_t* qrow = Q + (size_t)(qbase + (lane & 15)) * HDc;
#pragma unroll
  for (int c = 0; c < 4; ++c) aq[c] = load_a_frag(qrow, c * 32, lane);

  v8f o[8];
#pragma unroll
  for (int nc = 0; nc < 8; ++nc) o[nc] = (v8f){};
  float mi[8], li[8];
#pragma unroll
  for (int j = 0; j < 8; ++j) { mi[j] = -3.0e38f; li[j] = 0.f; }

  const float scale = 0.08838834764831845f;  // 1/sqrt(128)
  int kbmax = (qbase + 15) >> 5;             // inclusive

  for (int kb = 0; kb <= kbmax; ++kb) {
    int kstart = kb * 32;
    const half_t* krow0 = K + (size_t)(kstart + (lane & 15)) * HDc;
    const half_t* krow1 = krow0 + (size_t)16 * HDc;
    v8f c0 = {}, c1 = {};
#pragma unroll
    for (int c = 0; c < 4; ++c) {
      v16h b0 = load_b_frag(krow0, c * 32, lane);
      v16h b1 = load_b_frag(krow1, c * 32, lane);
      c0 = wmma_f16(aq[c], b0, c0);
      c1 = wmma_f16(aq[c], b1, c1);
    }

    int n0 = kstart + (lane & 15), n1 = n0 + 16;
    int mofs = (lane & 16) ? 8 : 0;
    float alpha[8];
#pragma unroll
    for (int j = 0; j < 8; ++j) {
      int mrow = qbase + j + mofs;
      float s0 = c0[j] * scale, s1 = c1[j] * scale;
      if (n0 > mrow) s0 = -3.0e38f;
      if (n1 > mrow) s1 = -3.0e38f;
      float rm = lane16_max(fmaxf(s0, s1));
      float mn = fmaxf(mi[j], rm);
      alpha[j] = expf(mi[j] - mn);
      mi[j] = mn;
      float p0 = expf(s0 - mn), p1 = expf(s1 - mn);
      float rs = lane16_sum(p0 + p1);
      li[j] = li[j] * alpha[j] + rs;
      sP[(j + mofs) * 32 + (lane & 15)]      = (half_t)p0;
      sP[(j + mofs) * 32 + 16 + (lane & 15)] = (half_t)p1;
    }

    // Stage V^T into LDS: lane = key index within block
    const half_t* vrow = V + (size_t)(kstart + lane) * HDc;
#pragma unroll
    for (int d = 0; d < HDc; d += 8) {
      v8h vv = *(const v8h*)(vrow + d);
#pragma unroll
      for (int t = 0; t < 8; ++t) sVt[(d + t) * 32 + lane] = vv[t];
    }
    __syncthreads();

#pragma unroll
    for (int nc = 0; nc < 8; ++nc)
#pragma unroll
      for (int j = 0; j < 8; ++j) o[nc][j] *= alpha[j];

    v16h ap = load_a_frag(sP + (size_t)(lane & 15) * 32, 0, lane);
#pragma unroll
    for (int nc = 0; nc < 8; ++nc) {
      v16h bv = load_b_frag(sVt + (size_t)(nc * 16 + (lane & 15)) * 32, 0, lane);
      o[nc] = wmma_f16(ap, bv, o[nc]);
    }
    __syncthreads();
  }

  int mofs = (lane & 16) ? 8 : 0;
#pragma unroll
  for (int nc = 0; nc < 8; ++nc) {
#pragma unroll
    for (int j = 0; j < 8; ++j) {
      int tok = b * Sc + qbase + j + mofs;
      int col = h * HDc + nc * 16 + (lane & 15);
      ctx16[(size_t)tok * Hc + col] = (half_t)(o[nc][j] / li[j]);
    }
  }
}

// ---------------------------------------------------------------------------
// residual add: resid2 = a + b ; out_init = resid2 (MoE accumulates on top)
// ---------------------------------------------------------------------------
__global__ __launch_bounds__(256) void add_resid_kernel(
    const float* __restrict__ a, const float* __restrict__ b,
    float* __restrict__ resid2, float* __restrict__ out_init, size_t n) {
  size_t i = (size_t)blockIdx.x * 256 + threadIdx.x;
  if (i < n) { float v = a[i] + b[i]; resid2[i] = v; out_init[i] = v; }
}

__global__ void zero_ints_kernel(int* p, int n) {
  if ((int)threadIdx.x < n) p[threadIdx.x] = 0;
}

// ---------------------------------------------------------------------------
// Router: softmax over 8 experts, top-2, renormalize, build gather lists.
// ---------------------------------------------------------------------------
__global__ __launch_bounds__(256) void router_kernel(
    const half_t* __restrict__ t16, const float* __restrict__ rw,
    float* __restrict__ coeff, int* __restrict__ gather,
    int* __restrict__ cnts) {
  int tok = blockIdx.x;
  int lane = threadIdx.x & 31, w = threadIdx.x >> 5;  // w = expert
  __shared__ float slog[8];
  float s = 0.f;
  for (int k = lane; k < Hc; k += 32)
    s += (float)t16[(size_t)tok * Hc + k] * rw[(size_t)k * Ec + w];
  s = wave_sum(s);
  if (lane == 0) slog[w] = s;
  __syncthreads();
  if (threadIdx.x == 0) {
    float mx = -3.0e38f;
    for (int e = 0; e < Ec; ++e) mx = fmaxf(mx, slog[e]);
    float p[Ec], sum = 0.f;
    for (int e = 0; e < Ec; ++e) { p[e] = expf(slog[e] - mx); sum += p[e]; }
    for (int e = 0; e < Ec; ++e) p[e] /= sum;
    int i0 = 0;
    for (int e = 1; e < Ec; ++e) if (p[e] > p[i0]) i0 = e;
    int i1 = (i0 == 0) ? 1 : 0;
    for (int e = 0; e < Ec; ++e)
      if (e != i0 && p[e] > p[i1]) i1 = e;
    float tot = p[i0] + p[i1];
    for (int e = 0; e < Ec; ++e) coeff[(size_t)tok * Ec + e] = 0.f;
    coeff[(size_t)tok * Ec + i0] = p[i0] / tot;
    coeff[(size_t)tok * Ec + i1] = p[i1] / tot;
    int s0 = atomicAdd(&cnts[i0], 1); gather[(size_t)i0 * Tc + s0] = tok;
    int s1 = atomicAdd(&cnts[i1], 1); gather[(size_t)i1 * Tc + s1] = tok;
  }
}

// ---------------------------------------------------------------------------
// MoE stage 1 (gathered rows): hbuf[p][I] = silu(t@w1[e]) * (t@w3[e]).
// Block tile 64(P) x 64(N) x 32(K), dual-B LDS pipeline, async staging.
// Wave (wm=w&3, wn=w>>2) owns 16x32 per matrix: 2+2 accumulators.
// ---------------------------------------------------------------------------
__global__ __launch_bounds__(256) void moe_gemm1_kernel(
    const half_t* __restrict__ t16, const half_t* __restrict__ w1t,
    const half_t* __restrict__ w3t, const int* __restrict__ gather,
    const int* __restrict__ cnts, int e, half_t* __restrict__ hbuf) {
  int cnt = cnts[e];
  if ((int)blockIdx.x * 64 >= cnt) return;
  __shared__ half_t sA[2][64 * BKP];
  __shared__ half_t sB1[2][64 * BKP];
  __shared__ half_t sB3[2][64 * BKP];
  int tid = threadIdx.x, lane = tid & 31, wave = tid >> 5;
  int wm = wave & 3, wn = wave >> 2;
  int pblk = blockIdx.x * 64, nblk = blockIdx.y * 64;

  int arow = tid >> 2, aseg = (tid & 3) * 8;
  int p = pblk + arow;
  int pc = (p < cnt) ? p : (cnt - 1);
  int tok = gather[(size_t)e * Tc + pc];
  const half_t* aSrc  = t16 + (size_t)tok * Hc + aseg;
  const half_t* b1Src = w1t + ((size_t)e * Ic + nblk + arow) * Hc + aseg;
  const half_t* b3Src = w3t + ((size_t)e * Ic + nblk + arow) * Hc + aseg;
  unsigned slot = (unsigned)(arow * BKP + aseg);
  unsigned aDst[2]  = {lds_off(&sA[0][slot]),  lds_off(&sA[1][slot])};
  unsigned b1Dst[2] = {lds_off(&sB1[0][slot]), lds_off(&sB1[1][slot])};
  unsigned b3Dst[2] = {lds_off(&sB3[0][slot]), lds_off(&sB3[1][slot])};
  auto stage = [&](int buf, int k0) {
    async_ld_b128(aDst[buf], aSrc + k0);
    async_ld_b128(b1Dst[buf], b1Src + k0);
    async_ld_b128(b3Dst[buf], b3Src + k0);
  };

  v8f zero = {};
  v8f acc1[2] = {zero, zero}, acc3[2] = {zero, zero};
  int ksteps = Hc >> 5;
  stage(0, 0);
  wait_async0();
  __syncthreads();
  for (int ks = 0; ks < ksteps; ++ks) {
    if (ks + 1 < ksteps) stage((ks + 1) & 1, (ks + 1) * 32);
    const half_t* aL = &sA[ks & 1][(wm * 16 + (lane & 15)) * BKP];
    v16h af = load_a_frag(aL, 0, lane);
#pragma unroll
    for (int nc = 0; nc < 2; ++nc) {
      int brow = (wn * 32 + nc * 16 + (lane & 15)) * BKP;
      acc1[nc] = wmma_f16(af, load_b_frag(&sB1[ks & 1][brow], 0, lane), acc1[nc]);
      acc3[nc] = wmma_f16(af, load_b_frag(&sB3[ks & 1][brow], 0, lane), acc3[nc]);
    }
    wait_async0();
    __syncthreads();
  }
  int n0 = nblk + wn * 32 + (lane & 15);
  int p0 = pblk + wm * 16 + ((lane & 16) ? 8 : 0);
#pragma unroll
  for (int nc = 0; nc < 2; ++nc)
#pragma unroll
    for (int j = 0; j < 8; ++j) {
      float x = acc1[nc][j];
      float g = x / (1.f + expf(-x));  // silu
      hbuf[(size_t)(p0 + j) * Ic + n0 + nc * 16] = (half_t)(g * acc3[nc][j]);
    }
}

// ---------------------------------------------------------------------------
// MoE stage 2: out[token] += coeff[token,e] * (hbuf @ w2[e]); scatter rows.
// Same pipeline as gemm_lds_kernel; experts serialized on stream -> plain +=.
// ---------------------------------------------------------------------------
__global__ __launch_bounds__(256) void moe_gemm2_kernel(
    const half_t* __restrict__ hbuf, const half_t* __restrict__ w2t,
    const int* __restrict__ gather, const int* __restrict__ cnts,
    const float* __restrict__ coeff, int e, float* __restrict__ out) {
  int cnt = cnts[e];
  if ((int)blockIdx.x * 64 >= cnt) return;
  __shared__ half_t sA[2][64 * BKP];
  __shared__ half_t sB[2][128 * BKP];
  int tid = threadIdx.x, lane = tid & 31, wave = tid >> 5;
  int wm = wave & 3, wn = wave >> 2;
  int pblk = blockIdx.x * 64, nblk = blockIdx.y * 128;

  int arow = tid >> 2, aseg = (tid & 3) * 8;
  const half_t* aSrc = hbuf + (size_t)(pblk + arow) * Ic + aseg;
  unsigned aDst[2] = {lds_off(&sA[0][arow * BKP + aseg]),
                      lds_off(&sA[1][arow * BKP + aseg])};
  const half_t* bSrc[2];
  unsigned bDst[2][2];
#pragma unroll
  for (int i = 0; i < 2; ++i) {
    int c = tid + i * 256, brow = c >> 2, bseg = (c & 3) * 8;
    bSrc[i] = w2t + ((size_t)e * Hc + nblk + brow) * Ic + bseg;
    bDst[0][i] = lds_off(&sB[0][brow * BKP + bseg]);
    bDst[1][i] = lds_off(&sB[1][brow * BKP + bseg]);
  }
  auto stage = [&](int buf, int k0) {
    async_ld_b128(aDst[buf], aSrc + k0);
    async_ld_b128(bDst[buf][0], bSrc[0] + k0);
    async_ld_b128(bDst[buf][1], bSrc[1] + k0);
  };

  v8f zero = {};
  v8f acc[4] = {zero, zero, zero, zero};
  int ksteps = Ic >> 5;
  stage(0, 0);
  wait_async0();
  __syncthreads();
  for (int ks = 0; ks < ksteps; ++ks) {
    if (ks + 1 < ksteps) stage((ks + 1) & 1, (ks + 1) * 32);
    const half_t* aL = &sA[ks & 1][(wm * 16 + (lane & 15)) * BKP];
    v16h af = load_a_frag(aL, 0, lane);
#pragma unroll
    for (int nc = 0; nc < 4; ++nc) {
      const half_t* bL = &sB[ks & 1][(wn * 64 + nc * 16 + (lane & 15)) * BKP];
      acc[nc] = wmma_f16(af, load_b_frag(bL, 0, lane), acc[nc]);
    }
    wait_async0();
    __syncthreads();
  }
  int n0 = nblk + wn * 64 + (lane & 15);
  int p0 = pblk + wm * 16 + ((lane & 16) ? 8 : 0);
#pragma unroll
  for (int nc = 0; nc < 4; ++nc)
#pragma unroll
    for (int j = 0; j < 8; ++j) {
      int pp = p0 + j;
      if (pp < cnt) {
        int tok = gather[(size_t)e * Tc + pp];
        float c = coeff[(size_t)tok * Ec + e];
        out[(size_t)tok * Hc + n0 + nc * 16] += c * acc[nc][j];
      }
    }
}

// ---------------------------------------------------------------------------
// Host orchestration
// ---------------------------------------------------------------------------
extern "C" void kernel_launch(void* const* d_in, const int* in_sizes, int n_in,
                              void* d_out, int out_size, void* d_ws,
                              size_t ws_size, hipStream_t stream) {
  const float* hidden   = (const float*)d_in[0];
  const int*   pos_ids  = (const int*)d_in[1];
  const float* rms1_w   = (const float*)d_in[2];
  const float* rms2_w   = (const float*)d_in[3];
  const float* q_w      = (const float*)d_in[4];
  const float* k_w      = (const float*)d_in[5];
  const float* v_w      = (const float*)d_in[6];
  const float* o_w      = (const float*)d_in[7];
  const float* router_w = (const float*)d_in[8];
  const float* w1       = (const float*)d_in[9];
  const float* w2       = (const float*)d_in[10];
  const float* w3       = (const float*)d_in[11];
  float* out = (float*)d_out;

  char* ws = (char*)d_ws;
  auto alloc = [&](size_t bytes) -> void* {
    void* p = (void*)ws;
    ws += (bytes + 255) & ~(size_t)255;
    return p;
  };
  const size_t TH = (size_t)Tc * Hc;
  half_t* qwT  = (half_t*)alloc((size_t)Hc * Hc * sizeof(half_t));
  half_t* kwT  = (half_t*)alloc((size_t)Hc * Hc * sizeof(half_t));
  half_t* vwT  = (half_t*)alloc((size_t)Hc * Hc * sizeof(half_t));
  half_t* owT  = (half_t*)alloc((size_t)Hc * Hc * sizeof(half_t));
  half_t* w1T  = (half_t*)alloc((size_t)Ec * Hc * Ic * sizeof(half_t));
  half_t* w3T  = (half_t*)alloc((size_t)Ec * Hc * Ic * sizeof(half_t));
  half_t* w2T  = (half_t*)alloc((size_t)Ec * Hc * Ic * sizeof(half_t));
  half_t* h1   = (half_t*)alloc(TH * sizeof(half_t));
  float*  qf   = (float*)alloc(TH * sizeof(float));
  float*  kf   = (float*)alloc(TH * sizeof(float));
  float*  vf   = (float*)alloc(TH * sizeof(float));
  half_t* q16  = (half_t*)alloc(TH * sizeof(half_t));
  half_t* k16  = (half_t*)alloc(TH * sizeof(half_t));
  half_t* v16  = (half_t*)alloc(TH * sizeof(half_t));
  half_t* ctx16= (half_t*)alloc(TH * sizeof(half_t));
  float*  attnf= (float*)alloc(TH * sizeof(float));
  float*  res2 = (float*)alloc(TH * sizeof(float));
  half_t* t2   = (half_t*)alloc(TH * sizeof(half_t));
  float*  coeff= (float*)alloc((size_t)Tc * Ec * sizeof(float));
  int*    gth  = (int*)alloc((size_t)Ec * Tc * sizeof(int));
  int*    cnts = (int*)alloc(256);
  half_t* hbuf = (half_t*)alloc((size_t)Tc * Ic * sizeof(half_t));

  dim3 tpb16(16, 16);
  // Weight conversion: [K][N] f32 -> [N][K] f16
  transpose_cast_kernel<<<dim3(Hc/16, Hc/16, 1), tpb16, 0, stream>>>(q_w, qwT, Hc, Hc);
  transpose_cast_kernel<<<dim3(Hc/16, Hc/16, 1), tpb16, 0, stream>>>(k_w, kwT, Hc, Hc);
  transpose_cast_kernel<<<dim3(Hc/16, Hc/16, 1), tpb16, 0, stream>>>(v_w, vwT, Hc, Hc);
  transpose_cast_kernel<<<dim3(Hc/16, Hc/16, 1), tpb16, 0, stream>>>(o_w, owT, Hc, Hc);
  transpose_cast_kernel<<<dim3(Hc/16, Ic/16, Ec), tpb16, 0, stream>>>(w1, w1T, Hc, Ic);
  transpose_cast_kernel<<<dim3(Hc/16, Ic/16, Ec), tpb16, 0, stream>>>(w3, w3T, Hc, Ic);
  transpose_cast_kernel<<<dim3(Ic/16, Hc/16, Ec), tpb16, 0, stream>>>(w2, w2T, Ic, Hc);

  // RMSNorm 1
  rmsnorm_kernel<<<Tc, 256, 0, stream>>>(hidden, rms1_w, h1, Hc);

  // QKV projections (tiled async-LDS GEMM)
  dim3 ggrid(Tc / 64, Hc / 128);
  gemm_lds_kernel<<<ggrid, 256, 0, stream>>>(h1, qwT, qf, Tc, Hc, Hc);
  gemm_lds_kernel<<<ggrid, 256, 0, stream>>>(h1, kwT, kf, Tc, Hc, Hc);
  gemm_lds_kernel<<<ggrid, 256, 0, stream>>>(h1, vwT, vf, Tc, Hc, Hc);

  // RoPE + permute
  rope_permute_kernel<<<dim3(Tc, NHc), 64, 0, stream>>>(qf, kf, pos_ids, q16, k16);
  v_permute_kernel<<<dim3(Tc, NHc), 128, 0, stream>>>(vf, v16);

  // Flash attention
  flash_attn_kernel<<<dim3(Sc / 16, NHc, Bc), 32, 0, stream>>>(q16, k16, v16, ctx16);

  // Output projection + residual
  gemm_lds_kernel<<<ggrid, 256, 0, stream>>>(ctx16, owT, attnf, Tc, Hc, Hc);
  add_resid_kernel<<<(unsigned)((TH + 255) / 256), 256, 0, stream>>>(hidden, attnf, res2, out, TH);

  // RMSNorm 2 + routing
  rmsnorm_kernel<<<Tc, 256, 0, stream>>>(res2, rms2_w, t2, Hc);
  zero_ints_kernel<<<1, 32, 0, stream>>>(cnts, Ec);
  router_kernel<<<Tc, 256, 0, stream>>>(t2, router_w, coeff, gth, cnts);

  // Routed experts: only top-2 work is done (gathered GEMMs)
  for (int e = 0; e < Ec; ++e) {
    moe_gemm1_kernel<<<dim3(Tc / 64, Ic / 64), 256, 0, stream>>>(t2, w1T, w3T, gth, cnts, e, hbuf);
    moe_gemm2_kernel<<<dim3(Tc / 64, Hc / 128), 256, 0, stream>>>(hbuf, w2T, gth, cnts, coeff, e, out);
  }
}
